// PCAM_8770323218577
// MI455X (gfx1250) — matrix operand order repaired and checked by reading, not compiled, not run on hardware
//
#include <hip/hip_runtime.h>
#include <hip/hip_bf16.h>

// ---------------------------------------------------------------------------
// Types for CDNA5 WMMA (wave32): V_WMMA_F32_16X16X32_BF16
// ---------------------------------------------------------------------------
typedef __attribute__((ext_vector_type(16))) __bf16 v16bf;
typedef __attribute__((ext_vector_type(8)))  float  v8f;
typedef __attribute__((ext_vector_type(8)))  unsigned short ush8;

__device__ __forceinline__ unsigned short f2bf(float f) {
    union { float f; unsigned int u; } cv; cv.f = f;
    unsigned int u = cv.u;
    // round-to-nearest-even bf16 truncation
    return (unsigned short)((u + 0x7FFFu + ((u >> 16) & 1u)) >> 16);
}

// A fragment: tile stored row-major [M][Kpad] (bf16 bits). Per ISA layout:
// lane = m + 16*hi; elements 0..7 <-> K = hi*8+e ; elements 8..15 <-> K = 16+hi*8+e
__device__ __forceinline__ v16bf load_frag_a(const unsigned short* t, int stride,
                                             int row0, int k0, int lane) {
    int m = lane & 15, hi = lane >> 4;
    const unsigned short* p = t + (row0 + m) * stride + k0;
    union { v16bf v; ush8 h[2]; } u;
    u.h[0] = *(const ush8*)(p + hi * 8);
    u.h[1] = *(const ush8*)(p + 16 + hi * 8);
    return u.v;
}

// B fragment: tile stored [N][Kpad]. Per ISA layout: lane = n + 16*hi;
// elements 0..7 <-> K = hi*8+e ; elements 8..15 <-> K = 16+hi*8+e
__device__ __forceinline__ v16bf load_frag_b(const unsigned short* t, int stride,
                                             int col0, int k0, int lane) {
    int n = lane & 15, hi = lane >> 4;
    const unsigned short* p = t + (col0 + n) * stride + k0;
    union { v16bf v; ush8 h[2]; } u;
    u.h[0] = *(const ush8*)(p + hi * 8);
    u.h[1] = *(const ush8*)(p + 16 + hi * 8);
    return u.v;
}

#define LPAD 72  // LDS row stride in half-words: 144B rows -> conflict-free b128 reads

// ---------------------------------------------------------------------------
// CAM energy: E[b] = X Xt,  X = x[b] as [512 x 9216] fp32 -> bf16 WMMA, fp32 acc
// grid (8, 8, B), block 256 (8 waves). Each block: 64x64 output tile, BK=64.
// ---------------------------------------------------------------------------
__global__ __launch_bounds__(256) void cam_energy_kernel(const float* __restrict__ x,
                                                         float* __restrict__ E) {
    __shared__ __align__(16) unsigned short As[64 * LPAD];
    __shared__ __align__(16) unsigned short Bs[64 * LPAD];
    const int b  = blockIdx.z;
    const int i0 = blockIdx.y * 64, j0 = blockIdx.x * 64;
    const float* Xb = x + (size_t)b * 512 * 9216;
    const int tid = threadIdx.x, lane = tid & 31, wave = tid >> 5;
    const int tm = wave >> 1, tn0 = (wave & 1) * 2;

    v8f c0 = {}; v8f c1 = {};
    for (int k0 = 0; k0 < 9216; k0 += 64) {
        // stage A (rows i0..i0+63) and B (rows j0..j0+63), fp32->bf16
        for (int p = tid; p < 1024; p += 256) {
            int r = p >> 4, cc = (p & 15) * 4;
            float4 va = *(const float4*)(Xb + (size_t)(i0 + r) * 9216 + k0 + cc);
            unsigned short* da = &As[r * LPAD + cc];
            da[0] = f2bf(va.x); da[1] = f2bf(va.y); da[2] = f2bf(va.z); da[3] = f2bf(va.w);
            float4 vb = *(const float4*)(Xb + (size_t)(j0 + r) * 9216 + k0 + cc);
            unsigned short* db = &Bs[r * LPAD + cc];
            db[0] = f2bf(vb.x); db[1] = f2bf(vb.y); db[2] = f2bf(vb.z); db[3] = f2bf(vb.w);
        }
        __syncthreads();
#pragma unroll
        for (int ks = 0; ks < 64; ks += 32) {
            v16bf a  = load_frag_a(As, LPAD, tm * 16, ks, lane);
            v16bf b0 = load_frag_b(Bs, LPAD, tn0 * 16, ks, lane);
            v16bf b1 = load_frag_b(Bs, LPAD, (tn0 + 1) * 16, ks, lane);
            c0 = __builtin_amdgcn_wmma_f32_16x16x32_bf16(false, a, false, b0, (short)0, c0, false, false);
            c1 = __builtin_amdgcn_wmma_f32_16x16x32_bf16(false, a, false, b1, (short)0, c1, false, false);
        }
        __syncthreads();
    }
    float* Eb = E + ((size_t)b * 512 + i0) * 512 + j0;
    const int n = lane & 15, hi = lane >> 4;
#pragma unroll
    for (int r = 0; r < 8; ++r) {
        int row = tm * 16 + r + 8 * hi;
        Eb[(size_t)row * 512 + tn0 * 16 + n]       = c0[r];
        Eb[(size_t)row * 512 + (tn0 + 1) * 16 + n] = c1[r];
    }
}

// ---------------------------------------------------------------------------
// CAM softmax: softmax(rowmax - E) == softmax(-E); stabilize with rowmin.
// One 256-thread block per row of 512. In-place.
// ---------------------------------------------------------------------------
__global__ __launch_bounds__(256) void cam_softmax_kernel(float* __restrict__ E) {
    float* e = E + (size_t)blockIdx.x * 512;
    const int tid = threadIdx.x, lane = tid & 31, wv = tid >> 5;
    __shared__ float red[8];
    float v0 = e[tid], v1 = e[tid + 256];
    float mn = fminf(v0, v1);
#pragma unroll
    for (int o = 16; o; o >>= 1) mn = fminf(mn, __shfl_xor(mn, o, 32));
    if (lane == 0) red[wv] = mn;
    __syncthreads();
    if (tid == 0) { float t = red[0]; for (int i = 1; i < 8; ++i) t = fminf(t, red[i]); red[0] = t; }
    __syncthreads();
    const float rmin = red[0];
    __syncthreads();
    float e0 = expf(rmin - v0), e1 = expf(rmin - v1);
    float s = e0 + e1;
#pragma unroll
    for (int o = 16; o; o >>= 1) s += __shfl_xor(s, o, 32);
    if (lane == 0) red[wv] = s;
    __syncthreads();
    if (tid == 0) { float t = red[0]; for (int i = 1; i < 8; ++i) t += red[i]; red[0] = t; }
    __syncthreads();
    const float inv = 1.0f / red[0];
    e[tid] = e0 * inv;
    e[tid + 256] = e1 * inv;
}

// ---------------------------------------------------------------------------
// CAM out: out[b, 512+c, n] = gamma * (attn @ X)[c,n] + x[b,c,n]
// grid (72, 8, B), block 256. 64(M) x 128(N) tile, K=512 in chunks of 64.
// X tile stored transposed in LDS ([N][K]) so B fragments are contiguous-K.
// ---------------------------------------------------------------------------
__global__ __launch_bounds__(256) void cam_out_kernel(const float* __restrict__ attn,
                                                      const float* __restrict__ x,
                                                      const float* __restrict__ gamma,
                                                      float* __restrict__ out) {
    __shared__ __align__(16) unsigned short As[64 * LPAD];
    __shared__ __align__(16) unsigned short Bs[128 * LPAD];
    const int b = blockIdx.z, m0 = blockIdx.y * 64, n0 = blockIdx.x * 128;
    const float* Ab = attn + (size_t)b * 512 * 512;
    const float* Xb = x + (size_t)b * 512 * 9216;
    const int tid = threadIdx.x, lane = tid & 31, wave = tid >> 5;
    const int tm = wave >> 1, tn0 = (wave & 1) * 4;

    v8f acc[4] = {};
    for (int k0 = 0; k0 < 512; k0 += 64) {
        for (int p = tid; p < 1024; p += 256) {           // attn tile [64 x 64]
            int r = p >> 4, cc = (p & 15) * 4;
            float4 v = *(const float4*)(Ab + (size_t)(m0 + r) * 512 + k0 + cc);
            unsigned short* d = &As[r * LPAD + cc];
            d[0] = f2bf(v.x); d[1] = f2bf(v.y); d[2] = f2bf(v.z); d[3] = f2bf(v.w);
        }
        for (int p = tid; p < 2048; p += 256) {           // X tile [64 x 128] -> Bs[n][k]
            int r = p >> 5, cc = (p & 31) * 4;
            float4 v = *(const float4*)(Xb + (size_t)(k0 + r) * 9216 + n0 + cc);
            Bs[(cc + 0) * LPAD + r] = f2bf(v.x);
            Bs[(cc + 1) * LPAD + r] = f2bf(v.y);
            Bs[(cc + 2) * LPAD + r] = f2bf(v.z);
            Bs[(cc + 3) * LPAD + r] = f2bf(v.w);
        }
        __syncthreads();
#pragma unroll
        for (int ks = 0; ks < 64; ks += 32) {
            v16bf a = load_frag_a(As, LPAD, tm * 16, ks, lane);
#pragma unroll
            for (int t = 0; t < 4; ++t) {
                v16bf bb = load_frag_b(Bs, LPAD, (tn0 + t) * 16, ks, lane);
                acc[t] = __builtin_amdgcn_wmma_f32_16x16x32_bf16(false, a, false, bb, (short)0, acc[t], false, false);
            }
        }
        __syncthreads();
    }
    const float g = gamma[0];
    const int n = lane & 15, hi = lane >> 4;
#pragma unroll
    for (int t = 0; t < 4; ++t) {
#pragma unroll
        for (int r = 0; r < 8; ++r) {
            int row = m0 + tm * 16 + r + 8 * hi;          // channel
            int col = n0 + (tn0 + t) * 16 + n;            // spatial
            float xv = Xb[(size_t)row * 9216 + col];
            out[((size_t)b * 1024 + 512 + row) * 9216 + col] = g * acc[t][r] + xv;
        }
    }
}

// ---------------------------------------------------------------------------
// BAM small kernels (sub-GFLOP, bandwidth-trivial)
// ---------------------------------------------------------------------------
__global__ __launch_bounds__(256) void bam_pool_kernel(const float* __restrict__ x,
                                                       float* __restrict__ xf) {
    int idx = blockIdx.x * 256 + threadIdx.x;             // b*512*144 + c*144 + n
    if (idx >= 4 * 512 * 144) return;
    int n = idx % 144, bc = idx / 144;
    int hy = n / 12, wx = n % 12;
    const float* p = x + (size_t)bc * 9216 + hy * 8 * 96 + wx * 8;
    float s = 0.f;
#pragma unroll
    for (int r = 0; r < 8; ++r) {
        float4 a = *(const float4*)(p + r * 96);
        float4 b = *(const float4*)(p + r * 96 + 4);
        s += a.x + a.y + a.z + a.w + b.x + b.y + b.z + b.w;
    }
    xf[idx] = s * (1.0f / 64.0f);
}

__global__ __launch_bounds__(256) void bam_qkv_kernel(const float* __restrict__ xf,
        const float* __restrict__ Wq, const float* __restrict__ bq,
        const float* __restrict__ Wk, const float* __restrict__ bk,
        const float* __restrict__ Wv, const float* __restrict__ bv,
        float* __restrict__ q, float* __restrict__ k, float* __restrict__ v) {
    int idx = blockIdx.x * 256 + threadIdx.x;             // 4 * 640 * 144
    if (idx >= 4 * 640 * 144) return;
    int n = idx % 144, t = idx / 144, o = t % 640, b = t / 640;
    const float* W; const float* bias; float* dst; int orow; int oc;
    if (o < 64)       { W = Wq; bias = bq; dst = q; orow = o;       oc = 64;  }
    else if (o < 128) { W = Wk; bias = bk; dst = k; orow = o - 64;  oc = 64;  }
    else              { W = Wv; bias = bv; dst = v; orow = o - 128; oc = 512; }
    const float* wr = W + (size_t)orow * 512;
    const float* xb = xf + (size_t)b * 512 * 144 + n;
    float s = bias[orow];
#pragma unroll 4
    for (int c = 0; c < 512; ++c) s += wr[c] * xb[(size_t)c * 144];
    dst[((size_t)b * oc + orow) * 144 + n] = s;
}

__global__ __launch_bounds__(256) void bam_energy_kernel(const float* __restrict__ q,
                                                         const float* __restrict__ k,
                                                         float* __restrict__ energy) {
    int idx = blockIdx.x * 256 + threadIdx.x;             // 4 * 144 * 144
    if (idx >= 4 * 144 * 144) return;
    int j = idx % 144, t = idx / 144, i = t % 144, b = t / 144;
    const float* qp = q + (size_t)b * 64 * 144 + i;
    const float* kp = k + (size_t)b * 64 * 144 + j;
    float s = 0.f;
#pragma unroll 4
    for (int c = 0; c < 64; ++c) s += qp[(size_t)c * 144] * kp[(size_t)c * 144];
    energy[idx] = s * 0.125f;                             // 64^-0.5
}

__global__ __launch_bounds__(256) void bam_softmax_kernel(float* __restrict__ energy) {
    int row = blockIdx.x * 8 + (threadIdx.x >> 5);        // 576 rows, wave per row
    int lane = threadIdx.x & 31;
    float* e = energy + (size_t)row * 144;
    float m = -3.402823466e+38f;
    for (int j = lane; j < 144; j += 32) m = fmaxf(m, e[j]);
#pragma unroll
    for (int o = 16; o; o >>= 1) m = fmaxf(m, __shfl_xor(m, o, 32));
    float s = 0.f;
    for (int j = lane; j < 144; j += 32) s += expf(e[j] - m);
#pragma unroll
    for (int o = 16; o; o >>= 1) s += __shfl_xor(s, o, 32);
    float inv = 1.0f / s;
    for (int j = lane; j < 144; j += 32) e[j] = expf(e[j] - m) * inv;
}

__global__ __launch_bounds__(256) void bam_apply_kernel(const float* __restrict__ v,
                                                        const float* __restrict__ attn,
                                                        float* __restrict__ bamout) {
    int idx = blockIdx.x * 256 + threadIdx.x;             // 4 * 512 * 144
    if (idx >= 4 * 512 * 144) return;
    int i = idx % 144, t = idx / 144, c = t % 512, b = t / 512;
    const float* vp = v + ((size_t)b * 512 + c) * 144;
    const float* ap = attn + ((size_t)b * 144 + i) * 144;
    float s = 0.f;
#pragma unroll 4
    for (int j = 0; j < 144; ++j) s += vp[j] * ap[j];
    bamout[idx] = s;
}

__global__ __launch_bounds__(256) void bam_upsample_kernel(const float* __restrict__ bamout,
                                                           const float* __restrict__ x,
                                                           float* __restrict__ out) {
    size_t idx = (size_t)blockIdx.x * 256 + threadIdx.x;  // 4 * 512 * 9216
    if (idx >= (size_t)4 * 512 * 9216) return;
    int hw = (int)(idx % 9216);
    size_t bc = idx / 9216;
    int b = (int)(bc / 512), c = (int)(bc % 512);
    int y = hw / 96, xx = hw % 96;
    float bv = bamout[bc * 144 + (y >> 3) * 12 + (xx >> 3)];
    out[((size_t)b * 1024 + c) * 9216 + hw] = bv + x[bc * 9216 + hw];
}

// ---------------------------------------------------------------------------
// Host launcher
// ---------------------------------------------------------------------------
extern "C" void kernel_launch(void* const* d_in, const int* in_sizes, int n_in,
                              void* d_out, int out_size, void* d_ws, size_t ws_size,
                              hipStream_t stream) {
    const float* x     = (const float*)d_in[0];
    const float* Wq    = (const float*)d_in[1];
    const float* bq    = (const float*)d_in[2];
    const float* Wk    = (const float*)d_in[3];
    const float* bk    = (const float*)d_in[4];
    const float* Wv    = (const float*)d_in[5];
    const float* bv    = (const float*)d_in[6];
    const float* gamma = (const float*)d_in[7];
    float* out = (float*)d_out;

    float* ws = (float*)d_ws;
    // workspace layout (floats)
    float* XF = ws;                          // 4*512*144  = 294912
    float* Q  = XF + 294912;                 // 4*64*144   = 36864
    float* K  = Q  + 36864;                  // 36864
    float* V  = K  + 36864;                  // 4*512*144  = 294912
    float* EN = V  + 294912;                 // 4*144*144  = 82944   (bam energy/attn)
    float* BO = EN + 82944;                  // 4*512*144  = 294912  (bam out, pooled res)
    float* CE = BO + 294912;                 // 4*512*512  = 1048576 (cam energy/attn)

    // ---- BAM path ----
    bam_pool_kernel<<<1152, 256, 0, stream>>>(x, XF);
    bam_qkv_kernel<<<1440, 256, 0, stream>>>(XF, Wq, bq, Wk, bk, Wv, bv, Q, K, V);
    bam_energy_kernel<<<324, 256, 0, stream>>>(Q, K, EN);
    bam_softmax_kernel<<<72, 256, 0, stream>>>(EN);
    bam_apply_kernel<<<1152, 256, 0, stream>>>(V, EN, BO);
    bam_upsample_kernel<<<73728, 256, 0, stream>>>(BO, x, out);

    // ---- CAM path (WMMA bf16) ----
    cam_energy_kernel<<<dim3(8, 8, 4), 256, 0, stream>>>(x, CE);
    cam_softmax_kernel<<<2048, 256, 0, stream>>>(CE);
    cam_out_kernel<<<dim3(72, 8, 4), 256, 0, stream>>>(CE, x, gamma, out);
}
